// NetV2_16681652977711
// MI455X (gfx1250) — matrix operand order repaired and checked by reading, not compiled, not run on hardware
//
#include <hip/hip_runtime.h>

#define EPS   1e-5f
#define BATCH 1024
#define HWD   28
#define PIX   (HWD * HWD)        // 784
#define NPIX  (BATCH * PIX)      // 802816
#define HW2   14
#define PIX2  (HW2 * HW2)        // 196
#define NPIX2 (BATCH * PIX2)     // 200704
#define C1    32
#define C2    64
#define C3    64
#define FC1K  (PIX2 * C3)        // 12544
#define FC1N  128
#define FC2N  10
#define KSTEPS (FC1K / 32)       // 392

typedef __attribute__((ext_vector_type(16))) _Float16 v16h;
typedef __attribute__((ext_vector_type(8)))  _Float16 v8h;
typedef __attribute__((ext_vector_type(8)))  float    v8f;

// ---------------------------------------------------------------------------
// Fragment-order ("swizzled") tiles: a 16x32 A-tile or 32x16 B-tile stored as
// 512 halves where lane L owns halves [L*16, L*16+16). This code's contiguous k order:
//   A (16Mx32K): lane = row + 16*(k>>4),     elem = k & 15
//   B (32Kx16N): lane = col + 16*(k>>4),     elem = k & 15
// Fragment load = one contiguous 32B read per lane (2x ds/global_load_b128).
// ---------------------------------------------------------------------------
__device__ __forceinline__ v16h frag_ld(const _Float16* tile, int lane) {
  return *(const v16h*)(tile + (lane << 4));
}

__device__ __forceinline__ v8f wmma_f16(v16h a, v16h b, v8f c) {
  return __builtin_amdgcn_wmma_f32_16x16x32_f16(false, a, false, b, (short)0, c,
                                                false, false);
}

// ---- Stage 0: zero dense+mask, scatter points -------------------------------
__global__ void k_zero(float* __restrict__ p, int n) {
  const int i = blockIdx.x * blockDim.x + threadIdx.x;
  if (i < n) p[i] = 0.0f;
}

__global__ void k_scatter(const float* __restrict__ feat, const int* __restrict__ idx,
                          float* __restrict__ dense, float* __restrict__ mask, int n) {
  const int i = blockIdx.x * blockDim.x + threadIdx.x;
  if (i >= n) return;
  const int b = idx[i * 3 + 0];
  const int y = idx[i * 3 + 1];
  const int x = idx[i * 3 + 2];
  const int p = (b * HWD + y) * HWD + x;
  atomicAdd(&dense[p], feat[i]);
  mask[p] = 1.0f;
}

// ---- Prep: fc1 weights f32 -> f16, B-fragment order in global ---------------
__global__ void k_prep_w(const float* __restrict__ w, _Float16* __restrict__ wsw) {
  const int i = blockIdx.x * blockDim.x + threadIdx.x;
  if (i >= FC1K * FC1N) return;
  const int K = i >> 7, j = i & 127;
  const int c = K & 63, pix = K >> 6;                 // NHWC k -> weight row
  const float v = w[(c * PIX2 + pix) * FC1N + j];
  const int k5 = K & 31;
  const size_t tile = (size_t)(K >> 5) * 8 + (j >> 4);
  wsw[tile * 512 + ((j & 15) + ((k5 >> 4) << 4)) * 16 + (k5 & 15)] = (_Float16)v;
}

// ---- Stage 1: conv 3x3 1->32 + BN + ReLU + mask (VALU, tiny) ----------------
__global__ void k_conv1(const float* __restrict__ dense, const float* __restrict__ mask,
                        const float* __restrict__ W1, const float* __restrict__ g,
                        const float* __restrict__ bb, const float* __restrict__ mn,
                        const float* __restrict__ vr, _Float16* __restrict__ act1) {
  const int p = blockIdx.x * blockDim.x + threadIdx.x;
  const int b = p / PIX, rem = p - b * PIX;
  const int y = rem / HWD, x = rem - y * HWD;
  float tp[9];
#pragma unroll
  for (int t = 0; t < 9; ++t) {
    const int yy = y + t / 3 - 1, xx = x + t % 3 - 1;
    tp[t] = (yy >= 0 && yy < HWD && xx >= 0 && xx < HWD)
              ? dense[(b * HWD + yy) * HWD + xx] : 0.0f;
  }
  const float mk = mask[p];
  __align__(16) _Float16 outv[C1];
#pragma unroll 4
  for (int c = 0; c < C1; ++c) {
    float acc = 0.0f;
#pragma unroll
    for (int t = 0; t < 9; ++t) acc += tp[t] * W1[t * C1 + c];
    const float s = g[c] * rsqrtf(vr[c] + EPS);
    float o = acc * s + (bb[c] - mn[c] * s);
    o = o > 0.0f ? o : 0.0f;
    outv[c] = (_Float16)(mk * o);
  }
#pragma unroll
  for (int j = 0; j < 4; ++j)
    *(v8h*)(act1 + (size_t)p * C1 + (j << 3)) = *(const v8h*)(outv + (j << 3));
}

// ---- Stage 2: conv 3x3 32->64, WMMA implicit GEMM (dominant ~30 GFLOP) ------
// 256 thr = 8 waves; wave w owns co-block (w&3), M-half (w>>2): 4 acc/wave.
// Each thread's staging rows/addresses are loop-invariant registers; A-stage
// is double-buffered so next tap's global loads overlap current tap's WMMAs.
__global__ void __launch_bounds__(256) k_conv2(
    const _Float16* __restrict__ act1, const float* __restrict__ mask,
    const float* __restrict__ W2, const float* __restrict__ g,
    const float* __restrict__ bb, const float* __restrict__ mn,
    const float* __restrict__ vr, _Float16* __restrict__ act2) {
  __shared__ __align__(32) _Float16 sW[36 * 512];       // 36 KB
  __shared__ __align__(32) _Float16 sA[2][8 * 512];     // 16 KB ping-pong
  __shared__ float sMk[128];
  const int t = threadIdx.x, lane = t & 31, wave = t >> 5;
  const int coBlk = wave & 3, mBase = (wave >> 2) << 2; // 0 or 4
  const int p0 = blockIdx.x * 128;
  for (int i = t; i < 9 * C1 * C2; i += 256) {          // weights, B-frag order
    const int tap = i >> 11, k = (i >> 6) & 31, co = i & 63;
    sW[((tap << 2) + (co >> 4)) * 512 +
       ((co & 15) + ((k >> 4) << 4)) * 16 + (k & 15)] = (_Float16)W2[i];
  }
  if (t < 128) sMk[t] = mask[p0 + t];
  // Loop-invariant staging metadata: thread t stages chunk (t&3) of rows
  // t>>2 and t>>2 + 64.
  const int r0 = t >> 2, jj = t & 3;
  int sy[2], sx[2], sb[2];
#pragma unroll
  for (int u = 0; u < 2; ++u) {
    const int p = p0 + r0 + (u << 6);
    const int b = p / PIX, rem = p - b * PIX;
    sy[u] = rem / HWD; sx[u] = rem - sy[u] * HWD;
    sb[u] = (b * HWD + sy[u]) * HWD + sx[u];
  }
  const int dOff = ((r0 >> 4) << 9) +
                   (((r0 & 15) + ((jj >> 1) << 4)) << 4) + ((jj & 1) << 3);
  auto stageA = [&](int tap, int buf) {
    const int dy = tap / 3 - 1, dx = tap % 3 - 1;
#pragma unroll
    for (int u = 0; u < 2; ++u) {
      v8h val = {};
      if ((unsigned)(sy[u] + dy) < HWD && (unsigned)(sx[u] + dx) < HWD)
        val = *(const v8h*)(act1 + (size_t)(sb[u] + dy * HWD + dx) * C1 + (jj << 3));
      *(v8h*)(sA[buf] + dOff + (u << 11)) = val;        // rows +64 -> tiles +4
    }
  };
  stageA(0, 0);
  v8f acc[4] = {};
  for (int tap = 0; tap < 9; ++tap) {
    __syncthreads();
    if (tap + 1 < 9) stageA(tap + 1, (tap + 1) & 1);
    const v16h fb = frag_ld(sW + (((tap << 2) + coBlk) << 9), lane);
#pragma unroll
    for (int mm = 0; mm < 4; ++mm) {
      const v16h fa = frag_ld(sA[tap & 1] + ((mBase + mm) << 9), lane);
      acc[mm] = wmma_f16(fa, fb, acc[mm]);
    }
  }
  const int co = (coBlk << 4) + (lane & 15);
  const float s   = g[co] * rsqrtf(vr[co] + EPS);
  const float bsh = bb[co] - mn[co] * s;
  const int rhi = (lane >> 4) << 3;
#pragma unroll
  for (int mm = 0; mm < 4; ++mm)
#pragma unroll
    for (int v = 0; v < 8; ++v) {
      const int r = ((mBase + mm) << 4) + rhi + v;
      float o = acc[mm][v] * s + bsh;
      o = fmaxf(o, 0.0f);
      act2[(size_t)(p0 + r) * C2 + co] = (_Float16)(sMk[r] * o);
    }
}

// ---- Stage 3: conv 2x2 s2 64->64, WMMA; writes fc1 A-fragment layout --------
__global__ void __launch_bounds__(256) k_conv3(
    const _Float16* __restrict__ act2, const float* __restrict__ mask,
    const float* __restrict__ W3, const float* __restrict__ g,
    const float* __restrict__ bb, const float* __restrict__ mn,
    const float* __restrict__ vr, _Float16* __restrict__ a_sw) {
  __shared__ __align__(32) _Float16 sW[32 * 512];       // 32 KB
  __shared__ __align__(32) _Float16 sA[16 * 512];       // 16 KB
  __shared__ int   sBG[128], sPix[128];
  __shared__ float sM[128];
  const int t = threadIdx.x, lane = t & 31, wave = t >> 5;
  const int coBlk = wave & 3, mBase = (wave >> 2) << 2;
  const int p0 = blockIdx.x * 128;
  for (int i = t; i < 4 * C2 * C3; i += 256) {          // weights, B-frag order
    const int tap = i >> 12, ci = (i >> 6) & 63, co = i & 63;
    const int k5 = ci & 31;
    sW[((((tap << 1) + (ci >> 5)) << 2) + (co >> 4)) * 512 +
       ((co & 15) + ((k5 >> 4) << 4)) * 16 + (k5 & 15)] = (_Float16)W3[i];
  }
  if (t < 128) {                                        // epilogue metadata
    const int p = p0 + t;
    const int b = p / PIX2, rem = p - b * PIX2;
    const int y = rem / HW2, x = rem - y * HW2;
    const int base = (b * HWD + 2 * y) * HWD + 2 * x;
    sBG[t] = b; sPix[t] = rem;
    const float* m = mask + base;                       // 2x2 max-pool of mask0
    sM[t] = fmaxf(fmaxf(m[0], m[1]), fmaxf(m[HWD], m[HWD + 1]));
  }
  // Loop-invariant staging metadata: thread t stages chunk (t&7) of rows
  // t>>3 + {0,32,64,96}.
  const int r0 = t >> 3, j = t & 7;
  const int h = j >> 2, jq = j & 3;
  int sb[4];
#pragma unroll
  for (int u = 0; u < 4; ++u) {
    const int p = p0 + r0 + (u << 5);
    const int b = p / PIX2, rem = p - b * PIX2;
    const int yy = rem / HW2, xx = rem - yy * HW2;
    sb[u] = (b * HWD + 2 * yy) * HWD + 2 * xx;
  }
  const int dOff = ((((r0 >> 4) << 1) + h) << 9) +
                   (((r0 & 15) + ((jq >> 1) << 4)) << 4) + ((jq & 1) << 3);
  v8f acc[4] = {};
  for (int tap = 0; tap < 4; ++tap) {
    const int dy = tap >> 1, dx = tap & 1;
    __syncthreads();
#pragma unroll
    for (int u = 0; u < 4; ++u) {                       // rows +32 -> tiles +4
      const v8h val =
          *(const v8h*)(act2 + (size_t)(sb[u] + dy * HWD + dx) * C2 + (j << 3));
      *(v8h*)(sA + dOff + (u << 11)) = val;
    }
    __syncthreads();
#pragma unroll
    for (int hh = 0; hh < 2; ++hh) {
      const v16h fb = frag_ld(sW + (((((tap << 1) + hh) << 2) + coBlk) << 9), lane);
#pragma unroll
      for (int mm = 0; mm < 4; ++mm) {
        const v16h fa = frag_ld(sA + ((((mBase + mm) << 1) + hh) << 9), lane);
        acc[mm] = wmma_f16(fa, fb, acc[mm]);
      }
    }
  }
  // Epilogue: BN+ReLU+mask1, store directly in fc1 A-fragment order.
  const int co = (coBlk << 4) + (lane & 15);
  const float s   = g[co] * rsqrtf(vr[co] + EPS);
  const float bsh = bb[co] - mn[co] * s;
  const int rhi   = (lane >> 4) << 3;
  const int lnOff = ((co >> 4) & 1) << 4;
  const int eOff  = co & 15;
#pragma unroll
  for (int mm = 0; mm < 4; ++mm)
#pragma unroll
    for (int v = 0; v < 8; ++v) {
      const int r = ((mBase + mm) << 4) + rhi + v;
      float o = acc[mm][v] * s + bsh;
      o = fmaxf(o, 0.0f);
      const int b = sBG[r], pix = sPix[r];
      const size_t tile = (size_t)(b >> 4) * KSTEPS + (pix << 1) + (co >> 5);
      a_sw[tile * 512 + ((b & 15) + lnOff) * 16 + eOff] = (_Float16)(sM[r] * o);
    }
}

// ---- Stage 4: FC 12544->128 + ReLU. LDS-free, both operands pre-swizzled ----
__global__ void __launch_bounds__(256) k_fc1(
    const _Float16* __restrict__ a_sw, const _Float16* __restrict__ w_sw,
    const float* __restrict__ bias, float* __restrict__ z) {
  const int t = threadIdx.x, lane = t & 31, wave = t >> 5;
  const int rg = blockIdx.x;
  const _Float16* ap = a_sw + (size_t)rg * KSTEPS * 512;
  v8f acc = {};
  for (int ks = 0; ks < KSTEPS; ks += 2) {
    if (ks + 2 < KSTEPS) {
      __builtin_prefetch(w_sw + (size_t)(((ks + 2) << 3) + wave) * 512 + (lane << 4), 0, 1);
      __builtin_prefetch(ap + (size_t)(ks + 2) * 512 + (lane << 4), 0, 1);
    }
    const v16h fa0 = frag_ld(ap + (size_t)ks * 512, lane);
    const v16h fb0 = frag_ld(w_sw + (size_t)((ks << 3) + wave) * 512, lane);
    acc = wmma_f16(fa0, fb0, acc);
    const v16h fa1 = frag_ld(ap + (size_t)(ks + 1) * 512, lane);
    const v16h fb1 = frag_ld(w_sw + (size_t)(((ks + 1) << 3) + wave) * 512, lane);
    acc = wmma_f16(fa1, fb1, acc);
  }
  const int j = (wave << 4) + (lane & 15);
  const float bj = bias[j];
  const int rhi = (lane >> 4) << 3;
#pragma unroll
  for (int v = 0; v < 8; ++v) {
    const int row = (rg << 4) + rhi + v;
    z[row * FC1N + j] = fmaxf(acc[v] + bj, 0.0f);
  }
}

// ---- Stage 5: FC 128->10 + log_softmax (tiny, VALU) -------------------------
__global__ void k_fc2(const float* __restrict__ z, const float* __restrict__ w,
                      const float* __restrict__ bias, float* __restrict__ out) {
  const int row = blockIdx.x * blockDim.x + threadIdx.x;
  if (row >= BATCH) return;
  float lg[FC2N];
#pragma unroll
  for (int j = 0; j < FC2N; ++j) lg[j] = bias[j];
  const float* zr = z + row * FC1N;
  for (int k = 0; k < FC1N; ++k) {
    const float zv = zr[k];
#pragma unroll
    for (int j = 0; j < FC2N; ++j) lg[j] += zv * w[k * FC2N + j];
  }
  float mx = lg[0];
#pragma unroll
  for (int j = 1; j < FC2N; ++j) mx = fmaxf(mx, lg[j]);
  float se = 0.0f;
#pragma unroll
  for (int j = 0; j < FC2N; ++j) se += expf(lg[j] - mx);
  const float lse = mx + logf(se);
#pragma unroll
  for (int j = 0; j < FC2N; ++j) out[row * FC2N + j] = lg[j] - lse;
}

// ---- Host launcher ----------------------------------------------------------
extern "C" void kernel_launch(void* const* d_in, const int* in_sizes, int n_in,
                              void* d_out, int out_size, void* d_ws, size_t ws_size,
                              hipStream_t stream) {
  const float* features = (const float*)d_in[0];
  const int*   indices  = (const int*)d_in[1];
  const float* W1 = (const float*)d_in[3];
  const float* g1 = (const float*)d_in[4];
  const float* b1 = (const float*)d_in[5];
  const float* m1 = (const float*)d_in[6];
  const float* v1 = (const float*)d_in[7];
  const float* W2 = (const float*)d_in[8];
  const float* g2 = (const float*)d_in[9];
  const float* b2 = (const float*)d_in[10];
  const float* m2 = (const float*)d_in[11];
  const float* v2 = (const float*)d_in[12];
  const float* W3 = (const float*)d_in[13];
  const float* g3 = (const float*)d_in[14];
  const float* b3 = (const float*)d_in[15];
  const float* m3 = (const float*)d_in[16];
  const float* v3 = (const float*)d_in[17];
  const float* fc1_w = (const float*)d_in[18];
  const float* fc1_b = (const float*)d_in[19];
  const float* fc2_w = (const float*)d_in[20];
  const float* fc2_b = (const float*)d_in[21];
  const int N = in_sizes[0];

  char* ws = (char*)d_ws;
  float*    dense = (float*)ws;     ws += (size_t)NPIX * 4;
  float*    mask  = (float*)ws;     ws += (size_t)NPIX * 4;
  _Float16* act1  = (_Float16*)ws;  ws += (size_t)NPIX * C1 * 2;
  _Float16* act2  = (_Float16*)ws;  ws += (size_t)NPIX * C2 * 2;
  _Float16* a_sw  = (_Float16*)ws;  ws += (size_t)NPIX2 * C3 * 2;   // fc1 A, frag order
  _Float16* w_sw  = (_Float16*)ws;  ws += (size_t)FC1K * FC1N * 2;  // fc1 B, frag order
  float*    zbuf  = (float*)ws;

  k_zero   <<<(2 * NPIX + 255) / 256,    256, 0, stream>>>(dense, 2 * NPIX);
  k_prep_w <<<(FC1K * FC1N + 255) / 256, 256, 0, stream>>>(fc1_w, w_sw);
  k_scatter<<<(N + 255) / 256,           256, 0, stream>>>(features, indices, dense, mask, N);
  k_conv1  <<<NPIX / 256,                256, 0, stream>>>(dense, mask, W1, g1, b1, m1, v1, act1);
  k_conv2  <<<NPIX / 128,                256, 0, stream>>>(act1, mask, W2, g2, b2, m2, v2, act2);
  k_conv3  <<<NPIX2 / 128,               256, 0, stream>>>(act2, mask, W3, g3, b3, m3, v3, a_sw);
  k_fc1    <<<BATCH / 16,                256, 0, stream>>>(a_sw, w_sw, fc1_b, zbuf);
  k_fc2    <<<BATCH / 256,               256, 0, stream>>>(zbuf, fc2_w, fc2_b, (float*)d_out);
}
